// GraphNetwork_4681514352874
// MI455X (gfx1250) — compile-verified, matrix operand order checked by reading, not run on hardware
//
#include <hip/hip_runtime.h>
#include <hip/hip_bf16.h>

// ---------------- problem constants (from reference) ----------------
#define NN      100000      // nodes
#define NE      1600000     // edges
#define NODE_D  128
#define EDGE_D  32
#define GLOB_D  16
#define HIDDEN  128
#define IN_D    208         // NODE_D + 2*EDGE_D + GLOB_D
#define K1PAD   224         // IN_D padded to multiple of 32 (WMMA K)
#define M_TILE  64          // nodes per block in the MLP kernel
#define XS_STR  232         // LDS stride (halves) for x tile: 464B = 29*16, 16B aligned
#define HS_STR  136         // LDS stride (halves) for h tile: 272B = 17*16, 16B aligned

typedef __attribute__((ext_vector_type(16))) _Float16 v16h;
typedef __attribute__((ext_vector_type(8)))  _Float16 v8h;
typedef __attribute__((ext_vector_type(8)))  float    v8f;

// Build a 16-half WMMA fragment from two contiguous 16-byte chunks:
// elements 0..7  = K = base..base+7        (VGPRs 0-3)
// elements 8..15 = K = base+16..base+23    (VGPRs 4-7)
// matching the CDNA5 16-bit A/B operand layout (lane's K sub-slice handled by caller).
__device__ __forceinline__ v16h make_frag(const _Float16* p) {
    v8h lo = *(const v8h*)(p);
    v8h hi = *(const v8h*)(p + 16);
    return __builtin_shufflevector(lo, hi, 0,1,2,3,4,5,6,7,8,9,10,11,12,13,14,15);
}

// ---------------- kernel 1: zero sent/recv accumulators ----------------
__global__ void zero_ws(float4* __restrict__ p, int n4) {
    int i = blockIdx.x * blockDim.x + threadIdx.x;
    if (i < n4) p[i] = make_float4(0.f, 0.f, 0.f, 0.f);
}

// ---------------- kernel 2: transpose + f16-convert weights ----------------
// Wt1[n][k] (k < K1PAD, zero padded past IN_D), Wt2[n][k]: N-major so a WMMA
// B-fragment per lane is two contiguous b128 loads.
__global__ void pack_weights(const float* __restrict__ W1, const float* __restrict__ W2,
                             _Float16* __restrict__ Wt1, _Float16* __restrict__ Wt2) {
    int tid = blockIdx.x * blockDim.x + threadIdx.x;
    if (tid < HIDDEN * K1PAD) {
        int n = tid / K1PAD, k = tid % K1PAD;
        float v = (k < IN_D) ? W1[k * HIDDEN + n] : 0.f;
        Wt1[n * K1PAD + k] = (_Float16)v;
    } else {
        int t = tid - HIDDEN * K1PAD;
        if (t < HIDDEN * HIDDEN) {
            int n = t / HIDDEN, k = t % HIDDEN;
            Wt2[n * HIDDEN + k] = (_Float16)W2[k * HIDDEN + n];
        }
    }
}

// ---------------- kernel 3: edge scatter (segment sums) ----------------
// tid = edge*32 + dim -> one wave = one edge, so sender/receiver index loads
// are wave-uniform; edge loads are fully coalesced; fp32 HW atomics into ws.
__global__ void edge_scatter(const float* __restrict__ edges,
                             const int*   __restrict__ snd,
                             const int*   __restrict__ rcv,
                             float* __restrict__ sent, float* __restrict__ recv) {
    int tid = blockIdx.x * blockDim.x + threadIdx.x;   // < NE*EDGE_D = 51.2M
    int dim = tid & (EDGE_D - 1);
    int e   = tid >> 5;
    float v = edges[tid];
    int s = snd[e];
    int r = rcv[e];
    __hip_atomic_fetch_add(&sent[(long)s * EDGE_D + dim], v,
                           __ATOMIC_RELAXED, __HIP_MEMORY_SCOPE_AGENT);
    __hip_atomic_fetch_add(&recv[(long)r * EDGE_D + dim], v,
                           __ATOMIC_RELAXED, __HIP_MEMORY_SCOPE_AGENT);
}

// ---------------- kernel 4: fused 2-layer node MLP via WMMA ----------------
__global__ __launch_bounds__(256)
void node_mlp(const float* __restrict__ nodes, const float* __restrict__ sent,
              const float* __restrict__ recv,  const float* __restrict__ glob,
              const _Float16* __restrict__ Wt1, const float* __restrict__ b1,
              const _Float16* __restrict__ Wt2, const float* __restrict__ b2,
              float* __restrict__ out) {
    __shared__ _Float16 xs[M_TILE * XS_STR];
    __shared__ _Float16 hs[M_TILE * HS_STR];

    const int tid   = threadIdx.x;
    const int node0 = blockIdx.x * M_TILE;

    // ---- stage x tile = [nodes | sent | recv | glob | 0-pad] as f16 in LDS
    // Trip count (64*224)/256 = 56 is uniform -> EXEC stays full for WMMA below.
    for (int idx = tid; idx < M_TILE * K1PAD; idx += 256) {
        int m = idx / K1PAD, k = idx % K1PAD;
        int node = node0 + m;
        float v = 0.f;
        if (node < NN) {
            if      (k < NODE_D)            v = nodes[(long)node * NODE_D + k];
            else if (k < NODE_D + EDGE_D)   v = sent[(long)node * EDGE_D + (k - NODE_D)];
            else if (k < NODE_D + 2*EDGE_D) v = recv[(long)node * EDGE_D + (k - NODE_D - EDGE_D)];
            else if (k < IN_D)              v = glob[k - (NODE_D + 2*EDGE_D)];
        }
        xs[m * XS_STR + k] = (_Float16)v;
    }
    __syncthreads();

    const int wave  = tid >> 5;
    const int lane  = tid & 31;
    const int rtile = (wave >> 1) * 16;        // this wave's 16-row strip
    const int chalf = (wave & 1) * 64;         // this wave's 64-col half
    const int mrow  = rtile + (lane & 15);     // A-fragment row for this lane
    const int kq    = (lane >> 4) * 8;         // lane's K sub-slice (0 or 8)
    const int ncol  = (lane & 15);             // B-fragment column-within-tile

    v8f zero = {};
    v8f acc[4] = {zero, zero, zero, zero};

    // ---- layer 1: h = relu(x @ W1 + b1), K = 224 (7 WMMA K-steps)
    for (int kb = 0; kb < K1PAD; kb += 32) {
        v16h a = make_frag(&xs[mrow * XS_STR + kb + kq]);
        #pragma unroll
        for (int t = 0; t < 4; ++t) {
            int n = chalf + t * 16 + ncol;
            v16h b = make_frag(&Wt1[n * K1PAD + kb + kq]);
            acc[t] = __builtin_amdgcn_wmma_f32_16x16x32_f16(
                false, a, false, b, (short)0, acc[t], false, false);
        }
    }
    {   // epilogue: bias + relu -> hs (f16). C/D layout: row = i + 8*(lane>=16).
        int rbase = rtile + 8 * (lane >> 4);
        #pragma unroll
        for (int t = 0; t < 4; ++t) {
            int col = chalf + t * 16 + ncol;
            float bias = b1[col];
            #pragma unroll
            for (int i = 0; i < 8; ++i) {
                float h = acc[t][i] + bias;
                hs[(rbase + i) * HS_STR + col] = (_Float16)(h > 0.f ? h : 0.f);
            }
            acc[t] = zero;   // reset for layer 2
        }
    }
    __syncthreads();

    // ---- layer 2: out = h @ W2 + b2, K = 128 (4 WMMA K-steps)
    for (int kb = 0; kb < HIDDEN; kb += 32) {
        v16h a = make_frag(&hs[mrow * HS_STR + kb + kq]);
        #pragma unroll
        for (int t = 0; t < 4; ++t) {
            int n = chalf + t * 16 + ncol;
            v16h b = make_frag(&Wt2[n * HIDDEN + kb + kq]);
            acc[t] = __builtin_amdgcn_wmma_f32_16x16x32_f16(
                false, a, false, b, (short)0, acc[t], false, false);
        }
    }
    {   // epilogue: bias + fp32 store (predicated AFTER all WMMA issue)
        int rbase = rtile + 8 * (lane >> 4);
        #pragma unroll
        for (int t = 0; t < 4; ++t) {
            int col = chalf + t * 16 + ncol;
            float bias = b2[col];
            #pragma unroll
            for (int i = 0; i < 8; ++i) {
                int node = node0 + rbase + i;
                if (node < NN)
                    out[(long)node * HIDDEN + col] = acc[t][i] + bias;
            }
        }
    }
}

// ---------------- host launcher ----------------
extern "C" void kernel_launch(void* const* d_in, const int* in_sizes, int n_in,
                              void* d_out, int out_size, void* d_ws, size_t ws_size,
                              hipStream_t stream) {
    const float* nodes     = (const float*)d_in[0];
    const float* edges     = (const float*)d_in[1];
    const int*   senders   = (const int*)  d_in[2];
    const int*   receivers = (const int*)  d_in[3];
    const float* glob      = (const float*)d_in[4];
    const float* W1        = (const float*)d_in[5];
    const float* b1        = (const float*)d_in[6];
    const float* W2        = (const float*)d_in[7];
    const float* b2        = (const float*)d_in[8];
    float* out = (float*)d_out;

    // workspace layout (all offsets 16B aligned)
    char* ws = (char*)d_ws;
    size_t sentBytes = (size_t)NN * EDGE_D * sizeof(float);       // 12.8 MB
    float*     sent = (float*)(ws);
    float*     recv = (float*)(ws + sentBytes);
    _Float16*  Wt1  = (_Float16*)(ws + 2 * sentBytes);            // 128*224 f16
    _Float16*  Wt2  = (_Float16*)(ws + 2 * sentBytes
                                     + (size_t)HIDDEN * K1PAD * sizeof(_Float16));

    // 1) zero sent+recv (contiguous 6.4M floats = 1.6M float4)
    int n4 = (NN * EDGE_D * 2) / 4;
    zero_ws<<<(n4 + 255) / 256, 256, 0, stream>>>((float4*)sent, n4);

    // 2) pack/transpose weights to f16 (tiny: 45056 elements)
    int wElems = HIDDEN * K1PAD + HIDDEN * HIDDEN;
    pack_weights<<<(wElems + 255) / 256, 256, 0, stream>>>(W1, W2, Wt1, Wt2);

    // 3) edge scatter: NE*32 = 51.2M threads, exact multiple of 256
    edge_scatter<<<(NE * EDGE_D) / 256, 256, 0, stream>>>(edges, senders, receivers,
                                                          sent, recv);

    // 4) fused MLP: 64 nodes per block
    node_mlp<<<(NN + M_TILE - 1) / M_TILE, 256, 0, stream>>>(
        nodes, sent, recv, glob, Wt1, b1, Wt2, b2, out);
}
